// SelfAttention_13606456394153
// MI455X (gfx1250) — matrix operand order repaired and checked by reading, not compiled, not run on hardware
//
#include <hip/hip_runtime.h>
#include <hip/hip_bf16.h>

// MI455X (gfx1250) single-head self-attention, B=4 S=2048 D=1024.
// All six GEMMs run through v_wmma_f32_16x16x32_bf16 (fp32 accumulate).
// K-panels are staged into double-buffered LDS with gfx1250 async copies
// (global_load_async_to_lds_b128, tracked on ASYNCcnt), overlapping the
// next panel's global traffic with the current panel's 16 WMMAs per wave.
// Workspace layout (104 MB total):
//   [0,16M)   Xb   bf16 [B*S, D]
//   [16,18M)  Wqb  bf16 [D, D]
//   [18,20M)  Wkb
//   [20,22M)  Wvb
//   [22,24M)  Wob
//   [24,40M)  Qb   bf16 [B*S, D]   (later reused as ctx)
//   [40,56M)  Kb   bf16 [B*S, D]
//   [56,72M)  Vtb  bf16 [B][D, S]  (V stored transposed per batch)
//   [72,104M) att  bf16 [B][S, S]  (softmax applied in place)

typedef __bf16 bf16x16 __attribute__((ext_vector_type(16)));
typedef float  f32x8   __attribute__((ext_vector_type(8)));

union Frag { uint4 u[2]; bf16x16 v; };

__device__ __forceinline__ unsigned short f32_to_bf16(float f) {
  unsigned int u = __float_as_uint(f);
  u += 0x7fffu + ((u >> 16) & 1u);           // round-to-nearest-even
  return (unsigned short)(u >> 16);
}
__device__ __forceinline__ float bf16_to_f32(unsigned short h) {
  return __uint_as_float(((unsigned int)h) << 16);
}

// gfx1250 async copy: global -> LDS, 16 bytes per lane, tracked on ASYNCcnt.
__device__ __forceinline__ void async_copy_b128(unsigned lds_addr, const void* gaddr) {
  asm volatile("global_load_async_to_lds_b128 %0, %1, off"
               :: "v"(lds_addr), "v"(gaddr) : "memory");
}
__device__ __forceinline__ void wait_async6() {
  asm volatile("s_wait_asynccnt 0x6" ::: "memory");
}
__device__ __forceinline__ void wait_async0() {
  asm volatile("s_wait_asynccnt 0x0" ::: "memory");
}

// ---------------------------------------------------------------- convert
__global__ __launch_bounds__(256)
void cvt_f32_bf16_kernel(const float* __restrict__ src,
                         unsigned short* __restrict__ dst, int n) {
  int i = (blockIdx.x * 256 + threadIdx.x) * 8;
  if (i >= n) return;
  const float4* s4 = (const float4*)(src + i);
  float4 a = s4[0], b = s4[1];
  unsigned short h[8] = {f32_to_bf16(a.x), f32_to_bf16(a.y),
                         f32_to_bf16(a.z), f32_to_bf16(a.w),
                         f32_to_bf16(b.x), f32_to_bf16(b.y),
                         f32_to_bf16(b.z), f32_to_bf16(b.w)};
  uint4 o;
  o.x = (unsigned)h[0] | ((unsigned)h[1] << 16);
  o.y = (unsigned)h[2] | ((unsigned)h[3] << 16);
  o.z = (unsigned)h[4] | ((unsigned)h[5] << 16);
  o.w = (unsigned)h[6] | ((unsigned)h[7] << 16);
  *(uint4*)(dst + i) = o;
}

// ---------------------------------------------------------------- WMMA GEMM
// C[M,N] = scale * (A[M,K] x B[N,K]^T) + bias[N]
// A, B bf16 row-major along K. Block tile 128x256, K-step 32, 8 waves in a
// 2x4 grid; each wave computes a 64x64 region (4x4 subtiles of 16x16 =
// 16 WMMAs per K-step). LDS panels double-buffered, filled by async copies.
template<bool OUT_F32, bool STORE_T>
__global__ __launch_bounds__(256)
void wmma_gemm_kernel(const unsigned short* __restrict__ A, int lda, long long sA,
                      const unsigned short* __restrict__ B, int ldb, long long sB,
                      const float* __restrict__ bias,
                      void* __restrict__ Cv, int ldc, long long sC,
                      int K, float scale)
{
  // pitch 40 bf16 (80 B) per 32-element row to spread LDS banks
  __shared__ unsigned short lA[2][128 * 40];   // 2 x 10 KB
  __shared__ unsigned short lB[2][256 * 40];   // 2 x 20 KB

  const int tid  = threadIdx.x;
  const int wave = tid >> 5;
  const int lane = tid & 31;
  const int l16  = lane & 15;
  const int hi   = lane >> 4;               // 0: lanes 0-15, 1: lanes 16-31
  const int wm   = wave >> 2;               // 0..1  -> 64-row slab
  const int wn   = wave & 3;                // 0..3  -> 64-col slab

  const int batch = blockIdx.z;
  const unsigned short* Ab = A + (long long)batch * sA
                               + (size_t)(blockIdx.x * 128) * lda;
  const unsigned short* Bb = B + (long long)batch * sB
                               + (size_t)(blockIdx.y * 256) * ldb;

  // staging assignment: A panel 128 rows x 32 elems -> 2 b128/thread;
  //                     B panel 256 rows x 32 elems -> 4 b128/thread
  const int arow = tid >> 1;                // 0..127
  const int acnk = (tid & 1) * 16;          // element offset 0 or 16
  const int brow = tid;                     // 0..255

  const unsigned ldsA = (unsigned)(uintptr_t)(&lA[0][0]);
  const unsigned ldsB = (unsigned)(uintptr_t)(&lB[0][0]);
  const unsigned aOff = (unsigned)((arow * 40 + acnk) * 2);
  const unsigned bOff = (unsigned)(brow * 40 * 2);

  f32x8 acc[4][4];
  #pragma unroll
  for (int mi = 0; mi < 4; ++mi)
    #pragma unroll
    for (int ni = 0; ni < 4; ++ni)
      acc[mi][ni] = (f32x8){0.f,0.f,0.f,0.f,0.f,0.f,0.f,0.f};

  const int nk = K >> 5;

  auto stage = [&](int kt, int b) {
    const unsigned short* ga = Ab + (size_t)arow * lda + kt * 32 + acnk;
    unsigned la = ldsA + (unsigned)b * (128u * 40u * 2u) + aOff;
    async_copy_b128(la,      ga);
    async_copy_b128(la + 16, ga + 8);
    const unsigned short* gb = Bb + (size_t)brow * ldb + kt * 32;
    unsigned lb = ldsB + (unsigned)b * (256u * 40u * 2u) + bOff;
    async_copy_b128(lb,      gb);
    async_copy_b128(lb + 16, gb + 8);
    async_copy_b128(lb + 32, gb + 16);
    async_copy_b128(lb + 48, gb + 24);
  };

  stage(0, 0);

  for (int kt = 0; kt < nk; ++kt) {
    const int buf = kt & 1;
    __syncthreads();                 // all waves done reading buf^1 (iter kt-1)
    if (kt + 1 < nk) {
      stage(kt + 1, buf ^ 1);        // next panel's copies fly during compute
      wait_async6();                 // oldest 6 (current panel) retired
    } else {
      wait_async0();
    }
    __syncthreads();                 // current panel visible to all waves

    // Fragments per ISA 16-bit layouts:
    //  A 16x32: lanes 0-15 hold K{0..7,16..23}, lanes 16-31 K{8..15,24..31}
    //  B 32x16: lanes 0-15 hold K{0..7,16..23}, lanes 16-31 K{8..15,24..31}
    Frag af[4], bfr[4];
    #pragma unroll
    for (int mi = 0; mi < 4; ++mi) {
      int r = wm * 64 + mi * 16 + l16;
      af[mi].u[0] = *(const uint4*)(&lA[buf][r * 40 + hi * 8]);
      af[mi].u[1] = *(const uint4*)(&lA[buf][r * 40 + 16 + hi * 8]);
    }
    #pragma unroll
    for (int ni = 0; ni < 4; ++ni) {
      int c = wn * 64 + ni * 16 + l16;
      bfr[ni].u[0] = *(const uint4*)(&lB[buf][c * 40 + hi * 8]);
      bfr[ni].u[1] = *(const uint4*)(&lB[buf][c * 40 + 16 + hi * 8]);
    }

    #pragma unroll
    for (int mi = 0; mi < 4; ++mi)
      #pragma unroll
      for (int ni = 0; ni < 4; ++ni)
        acc[mi][ni] = __builtin_amdgcn_wmma_f32_16x16x32_bf16(
            false, af[mi].v, false, bfr[ni].v,
            (short)0, acc[mi][ni], false, false);
  }

  // Epilogue. C/D layout: lanes 0-15 -> rows m0..m0+7 (vgpr j = row m0+j),
  // lanes 16-31 -> rows m0+8..m0+15; column = l16.
  #pragma unroll
  for (int ni = 0; ni < 4; ++ni) {
    const int gn = blockIdx.y * 256 + wn * 64 + ni * 16 + l16;
    const float bv = bias ? bias[gn] : 0.0f;
    #pragma unroll
    for (int mi = 0; mi < 4; ++mi) {
      const int gm0 = blockIdx.x * 128 + wm * 64 + mi * 16 + hi * 8;
      if (OUT_F32) {
        float* C = (float*)Cv + (long long)batch * sC;
        #pragma unroll
        for (int j = 0; j < 8; ++j)
          C[(size_t)(gm0 + j) * ldc + gn] = acc[mi][ni][j] * scale + bv;
      } else if (STORE_T) {
        unsigned short* C = (unsigned short*)Cv + (long long)batch * sC;
        unsigned short h[8];
        #pragma unroll
        for (int j = 0; j < 8; ++j)
          h[j] = f32_to_bf16(acc[mi][ni][j] * scale + bv);
        uint4 p;
        p.x = (unsigned)h[0] | ((unsigned)h[1] << 16);
        p.y = (unsigned)h[2] | ((unsigned)h[3] << 16);
        p.z = (unsigned)h[4] | ((unsigned)h[5] << 16);
        p.w = (unsigned)h[6] | ((unsigned)h[7] << 16);
        *(uint4*)(&C[(size_t)gn * ldc + gm0]) = p;   // 16B store along M
      } else {
        unsigned short* C = (unsigned short*)Cv + (long long)batch * sC;
        #pragma unroll
        for (int j = 0; j < 8; ++j)
          C[(size_t)(gm0 + j) * ldc + gn] = f32_to_bf16(acc[mi][ni][j] * scale + bv);
      }
    }
  }
}

// ---------------------------------------------------------------- softmax
// One 256-thread block per attention row (b*S+q); 8 elements per thread.
__global__ __launch_bounds__(256)
void softmax_mask_kernel(unsigned short* __restrict__ att,
                         const int* __restrict__ mask, int S)
{
  const int row  = blockIdx.x;            // = b*S + q; mask offset == row*S
  const int tid  = threadIdx.x;
  const int wave = tid >> 5;
  const int lane = tid & 31;
  unsigned short* arow = att + (size_t)row * S;
  const int*      mrow = mask + (size_t)row * S;
  const int i0 = tid * 8;

  uint4 av = *(const uint4*)(arow + i0);
  int4  m0 = *(const int4*)(mrow + i0);
  int4  m1 = *(const int4*)(mrow + i0 + 4);
  unsigned int w[4] = {av.x, av.y, av.z, av.w};
  int mk[8] = {m0.x, m0.y, m0.z, m0.w, m1.x, m1.y, m1.z, m1.w};

  float v[8];
  #pragma unroll
  for (int j = 0; j < 8; ++j) {
    unsigned short h = (unsigned short)((w[j >> 1] >> ((j & 1) * 16)) & 0xffffu);
    v[j] = (mk[j] == 0) ? -1e20f : bf16_to_f32(h);
  }

  float mx = v[0];
  #pragma unroll
  for (int j = 1; j < 8; ++j) mx = fmaxf(mx, v[j]);
  #pragma unroll
  for (int off = 16; off > 0; off >>= 1) mx = fmaxf(mx, __shfl_xor(mx, off, 32));

  __shared__ float red[8];
  if (lane == 0) red[wave] = mx;
  __syncthreads();
  float mall = red[0];
  #pragma unroll
  for (int k = 1; k < 8; ++k) mall = fmaxf(mall, red[k]);

  float e[8], s = 0.0f;
  #pragma unroll
  for (int j = 0; j < 8; ++j) { e[j] = __expf(v[j] - mall); s += e[j]; }
  #pragma unroll
  for (int off = 16; off > 0; off >>= 1) s += __shfl_xor(s, off, 32);
  __syncthreads();
  if (lane == 0) red[wave] = s;
  __syncthreads();
  float stot = 0.0f;
  #pragma unroll
  for (int k = 0; k < 8; ++k) stot += red[k];
  const float inv = 1.0f / stot;

  unsigned short h[8];
  #pragma unroll
  for (int j = 0; j < 8; ++j) h[j] = f32_to_bf16(e[j] * inv);
  uint4 p;
  p.x = (unsigned)h[0] | ((unsigned)h[1] << 16);
  p.y = (unsigned)h[2] | ((unsigned)h[3] << 16);
  p.z = (unsigned)h[4] | ((unsigned)h[5] << 16);
  p.w = (unsigned)h[6] | ((unsigned)h[7] << 16);
  *(uint4*)(arow + i0) = p;               // in place
}

// ---------------------------------------------------------------- launch
extern "C" void kernel_launch(void* const* d_in, const int* in_sizes, int n_in,
                              void* d_out, int out_size, void* d_ws, size_t ws_size,
                              hipStream_t stream)
{
  (void)in_sizes; (void)n_in; (void)out_size; (void)ws_size;
  const int B = 4, S = 2048, D = 1024;
  const float* X  = (const float*)d_in[0];
  const int*  msk = (const int*)d_in[1];
  const float* Wq = (const float*)d_in[2];
  const float* bq = (const float*)d_in[3];
  const float* Wk = (const float*)d_in[4];
  const float* bk = (const float*)d_in[5];
  const float* Wv = (const float*)d_in[6];
  const float* bv = (const float*)d_in[7];
  const float* Wo = (const float*)d_in[8];
  const float* bo = (const float*)d_in[9];
  float* out = (float*)d_out;

  char* ws = (char*)d_ws;
  const size_t MB = 1u << 20;
  unsigned short* Xb  = (unsigned short*)(ws);
  unsigned short* Wqb = (unsigned short*)(ws + 16 * MB);
  unsigned short* Wkb = (unsigned short*)(ws + 18 * MB);
  unsigned short* Wvb = (unsigned short*)(ws + 20 * MB);
  unsigned short* Wob = (unsigned short*)(ws + 22 * MB);
  unsigned short* Qb  = (unsigned short*)(ws + 24 * MB);  // reused as ctx
  unsigned short* Kb  = (unsigned short*)(ws + 40 * MB);
  unsigned short* Vtb = (unsigned short*)(ws + 56 * MB);
  unsigned short* att = (unsigned short*)(ws + 72 * MB);

  const long long SD = (long long)S * D;
  const long long SS = (long long)S * S;
  dim3 blk(256);

  // fp32 -> bf16
  cvt_f32_bf16_kernel<<<(B * S * D) / 2048, blk, 0, stream>>>(X, Xb, B * S * D);
  cvt_f32_bf16_kernel<<<(D * D) / 2048, blk, 0, stream>>>(Wq, Wqb, D * D);
  cvt_f32_bf16_kernel<<<(D * D) / 2048, blk, 0, stream>>>(Wk, Wkb, D * D);
  cvt_f32_bf16_kernel<<<(D * D) / 2048, blk, 0, stream>>>(Wv, Wvb, D * D);
  cvt_f32_bf16_kernel<<<(D * D) / 2048, blk, 0, stream>>>(Wo, Wob, D * D);

  // Q = X Wq^T + bq ; K = X Wk^T + bk   (M = B*S flattened)
  wmma_gemm_kernel<false, false><<<dim3(64, 4, 1), blk, 0, stream>>>(
      Xb, D, 0, Wqb, D, 0, bq, Qb, D, 0, D, 1.0f);
  wmma_gemm_kernel<false, false><<<dim3(64, 4, 1), blk, 0, stream>>>(
      Xb, D, 0, Wkb, D, 0, bk, Kb, D, 0, D, 1.0f);
  // V = X Wv^T + bv, stored transposed per batch: Vt[b] is [D, S]
  wmma_gemm_kernel<false, true><<<dim3(16, 4, 4), blk, 0, stream>>>(
      Xb, D, SD, Wvb, D, 0, bv, Vtb, S, SD, D, 1.0f);

  // att = (Q K^T) / sqrt(D)   -> bf16 [B][S, S]
  wmma_gemm_kernel<false, false><<<dim3(16, 8, 4), blk, 0, stream>>>(
      Qb, D, SD, Kb, D, SD, nullptr, att, S, SS, D, 0.03125f);

  // masked softmax, in place
  softmax_mask_kernel<<<B * S, blk, 0, stream>>>(att, msk, S);

  // ctx = P V   (B-operand = Vt[b], [N=D rows, K=S]) -> reuse Qb
  wmma_gemm_kernel<false, false><<<dim3(16, 4, 4), blk, 0, stream>>>(
      att, S, SS, Vtb, S, SD, nullptr, Qb, D, SD, S, 1.0f);

  // out = ctx Wo^T + bo  (fp32, M = B*S flattened)
  wmma_gemm_kernel<true, false><<<dim3(64, 4, 1), blk, 0, stream>>>(
      Qb, D, 0, Wob, D, 0, bo, out, D, 0, D, 1.0f);
}
